// ImprovedAttention_46110768890574
// MI455X (gfx1250) — matrix operand order repaired and checked by reading, not compiled, not run on hardware
//
#include <hip/hip_runtime.h>

// Problem constants: B=2, S=2048, E=2048, H=16, D=128, 3E=6144
#define BB   2
#define SS   2048
#define EE   2048
#define HH   16
#define DD   128
#define N3E  6144

typedef __attribute__((ext_vector_type(16))) __bf16       v16bf;
typedef __attribute__((ext_vector_type(8)))  float        v8f;
typedef __attribute__((ext_vector_type(4)))  unsigned int v4u;

union FragU { v16bf v; v4u q[2]; };

__device__ __forceinline__ v16bf load_frag(const unsigned short* p0,
                                           const unsigned short* p1) {
  FragU f;
  f.q[0] = *reinterpret_cast<const v4u*>(p0);
  f.q[1] = *reinterpret_cast<const v4u*>(p1);
  return f.v;
}

// Native f32<->bf16 converts (gfx1250 has bf16 ALU/cvt ops)
__device__ __forceinline__ unsigned short f2bf(float x) {
  union { __bf16 h; unsigned short u; } c; c.h = (__bf16)x; return c.u;
}
__device__ __forceinline__ float bf2f(unsigned short u) {
  union { unsigned short u; __bf16 h; } c; c.u = u; return (float)c.h;
}

#if __has_builtin(__builtin_amdgcn_exp2f)
#define EXP2F(x) __builtin_amdgcn_exp2f(x)
#else
#define EXP2F(x) exp2f(x)
#endif
#if __has_builtin(__builtin_amdgcn_rcpf)
#define RCPF(x) __builtin_amdgcn_rcpf(x)
#else
#define RCPF(x) (1.0f / (x))
#endif

#define WMMA_BF16(A, Bm, C) \
  __builtin_amdgcn_wmma_f32_16x16x32_bf16(false, (A), false, (Bm), (short)0, (C), false, false)

// ---------------------------------------------------------------------------
// Tensor Data Mover: async 2D tile load global -> LDS (D# per ISA 08 §8)
// ---------------------------------------------------------------------------
#if __has_builtin(__builtin_amdgcn_tensor_load_to_lds)
#define HAVE_TDM 1
typedef __attribute__((ext_vector_type(4))) unsigned int tdm_v4u;
typedef __attribute__((ext_vector_type(8))) int          tdm_v8i;
typedef __attribute__((ext_vector_type(4))) int          tdm_v4i;

// tile_w elements (2B each) per row, tile_h rows, row stride in elements.
__device__ __forceinline__ void tdm_load_2d(unsigned lds_addr, const void* gaddr,
                                            unsigned tile_w, unsigned tile_h,
                                            unsigned stride_elems) {
  unsigned long long ga = (unsigned long long)(size_t)gaddr;
  tdm_v4u g0;
  g0[0] = 1u;                                       // count=1 (valid user D#)
  g0[1] = lds_addr;                                 // LDS byte address
  g0[2] = (unsigned)(ga & 0xFFFFFFFFu);             // global_addr[31:0]
  g0[3] = (unsigned)((ga >> 32) & 0x01FFFFFFu)      // global_addr[56:32]
          | (2u << 30);                             // type=2 ("image")
  tdm_v8i g1;
  g1[0] = 0x00010000;                               // data_size=1 -> 2 bytes
  g1[1] = 0;                                        // tensor_dim0 = 1<<20 (lo16=0)
  g1[2] = 0x10;                                     // tensor_dim0 hi | tensor_dim1 lo
  g1[3] = (int)(0x10u | (tile_w << 16));            // tensor_dim1 hi | tile_dim0
  g1[4] = (int)tile_h;                              // tile_dim1 | tile_dim2=0
  g1[5] = (int)stride_elems;                        // tensor_dim0_stride[31:0]
  g1[6] = 0;                                        // stride0 hi | stride1 lo
  g1[7] = 0;                                        // stride1 hi
  tdm_v4i gz = {0, 0, 0, 0};
#if defined(__clang_major__) && (__clang_major__ >= 23)
  tdm_v8i z8 = {0, 0, 0, 0, 0, 0, 0, 0};
  __builtin_amdgcn_tensor_load_to_lds(g0, g1, gz, gz, z8, 0);
#else
  __builtin_amdgcn_tensor_load_to_lds(g0, g1, gz, gz, 0);
#endif
}
#else
#define HAVE_TDM 0
#endif

// ---------------------------------------------------------------------------
// 1) x (f32) -> x_bf16
// ---------------------------------------------------------------------------
__global__ __launch_bounds__(256) void cvt_x_kernel(const float* __restrict__ x,
                                                    unsigned short* __restrict__ xb,
                                                    int n) {
  int i = blockIdx.x * 256 + threadIdx.x;
  if (i < n) xb[i] = f2bf(x[i]);
}

// ---------------------------------------------------------------------------
// 2) W (E x 3E, f32) -> Wt_bf16 (3E x E): LDS-tiled transpose+convert
// ---------------------------------------------------------------------------
__global__ __launch_bounds__(256) void transpose_w_kernel(const float* __restrict__ w,
                                                          unsigned short* __restrict__ wt) {
  __shared__ float tile[32][33];
  const int n0 = blockIdx.x * 32;   // N (3E) tile
  const int k0 = blockIdx.y * 32;   // K (E) tile
  const int tx = threadIdx.x, ty = threadIdx.y;  // 32 x 8
#pragma unroll
  for (int j = 0; j < 32; j += 8)
    tile[ty + j][tx] = w[(size_t)(k0 + ty + j) * N3E + n0 + tx];
  __syncthreads();
#pragma unroll
  for (int j = 0; j < 32; j += 8)
    wt[(size_t)(n0 + ty + j) * EE + k0 + tx] = f2bf(tile[tx][ty + j]);
}

// ---------------------------------------------------------------------------
// 3) QKV GEMM: qkv_bf16(4096 x 6144) = x_bf16(4096 x 2048) * Wt^T
//    One wave per 64(M) x 32(N) tile; software-pipelined double-buffered
//    fragments over a 64-wide K step + global_prefetch 128 elements ahead.
// ---------------------------------------------------------------------------
__global__ __launch_bounds__(32) void gemm_qkv_kernel(const unsigned short* __restrict__ xb,
                                                      const unsigned short* __restrict__ wt,
                                                      unsigned short* __restrict__ qkv) {
  const int n0   = blockIdx.x * 32;
  const int m0   = blockIdx.y * 64;
  const int lane = threadIdx.x;
  const int hl   = lane >> 4;        // lane half
  const int l15  = lane & 15;
  const int ka   = hl * 16;          // A-operand K start (16 contiguous)
  const int kb   = hl * 16;          // B-operand K start (16 contiguous)

  const unsigned short* aptr[4];
  const unsigned short* bptr[2];
#pragma unroll
  for (int mt = 0; mt < 4; ++mt)
    aptr[mt] = xb + (size_t)(m0 + mt * 16 + l15) * EE + ka;
#pragma unroll
  for (int nt = 0; nt < 2; ++nt)
    bptr[nt] = wt + (size_t)(n0 + nt * 16 + l15) * EE + kb;

  v8f acc[4][2];
#pragma unroll
  for (int mt = 0; mt < 4; ++mt)
#pragma unroll
    for (int nt = 0; nt < 2; ++nt)
#pragma unroll
      for (int r = 0; r < 8; ++r) acc[mt][nt][r] = 0.0f;

  v16bf a0[4], b0[2], a1[4], b1[2];
#pragma unroll
  for (int mt = 0; mt < 4; ++mt) a0[mt] = load_frag(aptr[mt], aptr[mt] + 8);
#pragma unroll
  for (int nt = 0; nt < 2; ++nt) b0[nt] = load_frag(bptr[nt], bptr[nt] + 8);

#pragma unroll 1
  for (int kk = 0; kk < EE; kk += 64) {
    if (kk + 128 < EE) {
#pragma unroll
      for (int mt = 0; mt < 4; ++mt) __builtin_prefetch(aptr[mt] + kk + 128, 0, 1);
#pragma unroll
      for (int nt = 0; nt < 2; ++nt) __builtin_prefetch(bptr[nt] + kk + 128, 0, 1);
    }
    // stage 1 loads (kk+32) issue before stage 0 WMMAs
#pragma unroll
    for (int mt = 0; mt < 4; ++mt) {
      const unsigned short* p = aptr[mt] + kk + 32;
      a1[mt] = load_frag(p, p + 8);
    }
#pragma unroll
    for (int nt = 0; nt < 2; ++nt) {
      const unsigned short* p = bptr[nt] + kk + 32;
      b1[nt] = load_frag(p, p + 8);
    }
#pragma unroll
    for (int mt = 0; mt < 4; ++mt)
#pragma unroll
      for (int nt = 0; nt < 2; ++nt)
        acc[mt][nt] = WMMA_BF16(a0[mt], b0[nt], acc[mt][nt]);
    if (kk + 64 < EE) {
#pragma unroll
      for (int mt = 0; mt < 4; ++mt) {
        const unsigned short* p = aptr[mt] + kk + 64;
        a0[mt] = load_frag(p, p + 8);
      }
#pragma unroll
      for (int nt = 0; nt < 2; ++nt) {
        const unsigned short* p = bptr[nt] + kk + 64;
        b0[nt] = load_frag(p, p + 8);
      }
    }
#pragma unroll
    for (int mt = 0; mt < 4; ++mt)
#pragma unroll
      for (int nt = 0; nt < 2; ++nt)
        acc[mt][nt] = WMMA_BF16(a1[mt], b1[nt], acc[mt][nt]);
  }

#pragma unroll
  for (int mt = 0; mt < 4; ++mt)
#pragma unroll
    for (int nt = 0; nt < 2; ++nt)
#pragma unroll
      for (int r = 0; r < 8; ++r)
        qkv[(size_t)(m0 + mt * 16 + r + 8 * hl) * N3E + n0 + nt * 16 + l15] =
            f2bf(acc[mt][nt][r]);
}

// ---------------------------------------------------------------------------
// 4) In-place RoPE on q and k halves of qkv_bf16.
// ---------------------------------------------------------------------------
__global__ __launch_bounds__(256) void rope_kernel(unsigned short* __restrict__ qkv) {
  int idx = blockIdx.x * 256 + threadIdx.x;     // BB*SS*HH*64 = 4194304
  if (idx >= BB * SS * HH * 64) return;
  const int i = idx & 63;
  const int h = (idx >> 6) & (HH - 1);
  const int t = idx >> 10;                      // token id 0..4095
  const int s = t & (SS - 1);                   // position in sequence
  const size_t base = (size_t)t * N3E + h * DD;

  const float invf = __powf(10000.0f, -(float)i * (1.0f / 64.0f));
  const float ang  = (float)s * invf;
  float sn, cs;
  __sincosf(ang, &sn, &cs);

  {  // q
    unsigned short* p = qkv + base;
    float a = bf2f(p[i]), b = bf2f(p[i + 64]);
    p[i]      = f2bf(a * cs - b * sn);
    p[i + 64] = f2bf(b * cs + a * sn);
  }
  {  // k
    unsigned short* p = qkv + base + EE;
    float a = bf2f(p[i]), b = bf2f(p[i + 64]);
    p[i]      = f2bf(a * cs - b * sn);
    p[i + 64] = f2bf(b * cs + a * sn);
  }
}

// ---------------------------------------------------------------------------
// 5) V transpose per head: vt[(bh*D + d)*S + s] = qkv_v[b, s, h, d]
// ---------------------------------------------------------------------------
__global__ __launch_bounds__(256) void vtr_kernel(const unsigned short* __restrict__ qkv,
                                                  unsigned short* __restrict__ vt) {
  __shared__ unsigned short tile[32][33];
  const int s0 = blockIdx.x * 32;
  const int d0 = blockIdx.y * 32;
  const int bh = blockIdx.z;
  const int b  = bh >> 4, h = bh & 15;
  const int tx = threadIdx.x, ty = threadIdx.y;  // 32 x 8
#pragma unroll
  for (int j = 0; j < 32; j += 8)
    tile[ty + j][tx] =
        qkv[(size_t)(b * SS + s0 + ty + j) * N3E + 2 * EE + h * DD + d0 + tx];
  __syncthreads();
#pragma unroll
  for (int j = 0; j < 32; j += 8)
    vt[((size_t)bh * DD + d0 + ty + j) * SS + s0 + tx] = tile[tx][ty + j];
}

// ---------------------------------------------------------------------------
// 6) Flash attention: one wave per (head, 16-row q tile).
//    K/V tiles staged into double-buffered LDS by the Tensor Data Mover
//    (async, s_wait_tensorcnt), fragments via ds_load_b128; QK^T and PV via
//    16x16x32 bf16 WMMA; log2-domain online softmax.
// ---------------------------------------------------------------------------
__global__ __launch_bounds__(32) void attn_kernel(const unsigned short* __restrict__ qkv,
                                                  const unsigned short* __restrict__ vt,
                                                  float* __restrict__ out) {
  const int qb   = blockIdx.x;           // 0..127 (16-row q tiles)
  const int bh   = blockIdx.y;           // 0..31
  const int b    = bh >> 4, h = bh & 15;
  const int lane = threadIdx.x;
  const int hl   = lane >> 4;
  const int l15  = lane & 15;
  const int q0   = qb * 16;
  const int ka   = hl * 16;
  // log2(e) / sqrt(128): scores go straight to log2 domain
  const float c2 = 1.4426950408889634f * 0.08838834764831845f;

  // Q fragments: 4 chunks of 16x32
  const unsigned short* qbase = qkv + (size_t)(b * SS + q0 + l15) * N3E + h * DD;
  v16bf qf[4];
#pragma unroll
  for (int c = 0; c < 4; ++c)
    qf[c] = load_frag(qbase + c * 32 + ka, qbase + c * 32 + ka + 8);

  v8f o[8];
#pragma unroll
  for (int dc = 0; dc < 8; ++dc)
#pragma unroll
    for (int r = 0; r < 8; ++r) o[dc][r] = 0.0f;

  float m[8], l[8];
#pragma unroll
  for (int r = 0; r < 8; ++r) { m[r] = -1.0e30f; l[r] = 0.0f; }

  const unsigned short* kbase = qkv + (size_t)(b * SS) * N3E + EE + h * DD;
  const unsigned short* vtb   = vt + (size_t)bh * DD * SS;

  __shared__ __align__(16) unsigned short pbuf[16 * 32];

  const int nj = (q0 + 16 + 31) >> 5;   // 32-wide key blocks covering j <= q0+15

#if HAVE_TDM
  __shared__ __align__(16) unsigned short kbuf[2][32 * 128];  // 32 tokens x 128 d
  __shared__ __align__(16) unsigned short vbuf[2][128 * 32];  // 128 d x 32 tokens
  // Prologue: async-stage block 0 (K tile: 128 wide x 32 rows, stride 3E;
  //                                V tile: 32 wide x 128 rows, stride S)
  tdm_load_2d((unsigned)(size_t)&kbuf[0][0], kbase, 128, 32, N3E);
  tdm_load_2d((unsigned)(size_t)&vbuf[0][0], vtb, 32, 128, SS);
#endif

#pragma unroll 1
  for (int jb = 0; jb < nj; ++jb) {
    const int j0 = jb * 32;

#if HAVE_TDM
    if (jb + 1 < nj) {   // issue next block's DMAs before waiting on current
      const int jn = j0 + 32, nb = (jb + 1) & 1;
      tdm_load_2d((unsigned)(size_t)&kbuf[nb][0], kbase + (size_t)jn * N3E, 128, 32, N3E);
      tdm_load_2d((unsigned)(size_t)&vbuf[nb][0], vtb + jn, 32, 128, SS);
      __builtin_amdgcn_s_wait_tensorcnt(2);   // drain only the older pair
    } else {
      __builtin_amdgcn_s_wait_tensorcnt(0);
    }
    __asm__ volatile("" ::: "memory");        // fence LDS reads behind the wait
    const int buf = jb & 1;
#else
    if (jb + 1 < nj) {  // prefetch next key/value block
      __builtin_prefetch(kbase + (size_t)(j0 + 32 + l15) * N3E + hl * 16, 0, 1);
      __builtin_prefetch(kbase + (size_t)(j0 + 48 + l15) * N3E + hl * 16, 0, 1);
#pragma unroll
      for (int g = 0; g < 4; ++g)
        __builtin_prefetch(vtb + (size_t)(g * 32 + lane) * SS + j0 + 32, 0, 1);
    }
#endif

    float sv[2][8];
#pragma unroll
    for (int t = 0; t < 2; ++t) {
      const int jt = j0 + t * 16;
      v8f s;
#pragma unroll
      for (int r = 0; r < 8; ++r) s[r] = 0.0f;
#pragma unroll
      for (int c = 0; c < 4; ++c) {
#if HAVE_TDM
        const unsigned short* kp = &kbuf[buf][(t * 16 + l15) * 128 + c * 32 + hl * 16];
#else
        const unsigned short* kp =
            kbase + (size_t)(jt + l15) * N3E + c * 32 + hl * 16;
#endif
        v16bf kf = load_frag(kp, kp + 8);
        s = WMMA_BF16(qf[c], kf, s);
      }
      const int colg = jt + l15;
#pragma unroll
      for (int r = 0; r < 8; ++r) {
        const int rowg = q0 + r + 8 * hl;
        const float x = s[r] * c2;                   // log2-domain score
        sv[t][r] = (colg <= rowg) ? x : -1.0e30f;    // causal mask
      }
    }

    // Online softmax update (row reductions across each 16-lane half)
    float alpha[8];
#pragma unroll
    for (int r = 0; r < 8; ++r) {
      float mt = fmaxf(sv[0][r], sv[1][r]);
#pragma unroll
      for (int msk = 1; msk < 16; msk <<= 1)
        mt = fmaxf(mt, __shfl_xor(mt, msk, 32));
      const float mn = fmaxf(m[r], mt);
      alpha[r] = EXP2F(m[r] - mn);
      m[r] = mn;
      sv[0][r] = EXP2F(sv[0][r] - mn);
      sv[1][r] = EXP2F(sv[1][r] - mn);
      // sum what will actually be multiplied (bf16-rounded P)
      float rs = bf2f(f2bf(sv[0][r])) + bf2f(f2bf(sv[1][r]));
#pragma unroll
      for (int msk = 1; msk < 16; msk <<= 1)
        rs += __shfl_xor(rs, msk, 32);
      l[r] = l[r] * alpha[r] + rs;
    }

    // Restage P (C layout) -> LDS -> A-operand layout
    __syncthreads();
#pragma unroll
    for (int t = 0; t < 2; ++t)
#pragma unroll
      for (int r = 0; r < 8; ++r)
        pbuf[(r + 8 * hl) * 32 + t * 16 + l15] = f2bf(sv[t][r]);
    __syncthreads();
    const v16bf pf = load_frag(pbuf + l15 * 32 + ka, pbuf + l15 * 32 + ka + 8);

    // Rescale running O and accumulate P @ V
#pragma unroll
    for (int dc = 0; dc < 8; ++dc) {
#pragma unroll
      for (int r = 0; r < 8; ++r) o[dc][r] *= alpha[r];
#if HAVE_TDM
      const unsigned short* vp = &vbuf[buf][(dc * 16 + l15) * 32 + hl * 16];
#else
      const unsigned short* vp =
          vtb + (size_t)(dc * 16 + l15) * SS + j0 + hl * 16;
#endif
      v16bf vf = load_frag(vp, vp + 8);
      o[dc] = WMMA_BF16(pf, vf, o[dc]);
    }
  }

  // Normalize and write fp32 output: out[b, q0+row, h*D + d]
#pragma unroll
  for (int r = 0; r < 8; ++r) l[r] = RCPF(l[r]);
#pragma unroll
  for (int dc = 0; dc < 8; ++dc)
#pragma unroll
    for (int r = 0; r < 8; ++r)
      out[(size_t)(b * SS + q0 + r + 8 * hl) * EE + h * DD + dc * 16 + l15] =
          o[dc][r] * l[r];
}

// ---------------------------------------------------------------------------
// Launch
// ---------------------------------------------------------------------------
extern "C" void kernel_launch(void* const* d_in, const int* in_sizes, int n_in,
                              void* d_out, int out_size, void* d_ws, size_t ws_size,
                              hipStream_t stream) {
  const float* x = (const float*)d_in[0];
  // d_in[1] = mask: causal, handled analytically
  const float* w = (const float*)d_in[2];
  float* out = (float*)d_out;

  // Workspace layout (bf16 elements): total ~109 MB
  unsigned short* xb  = (unsigned short*)d_ws;                 // 4096*2048
  unsigned short* wt  = xb + (size_t)BB * SS * EE;             // 6144*2048
  unsigned short* qkv = wt + (size_t)EE * N3E;                 // 4096*6144
  unsigned short* vt  = qkv + (size_t)BB * SS * N3E;           // 32*128*2048

  const int nx = BB * SS * EE;
  cvt_x_kernel<<<(nx + 255) / 256, 256, 0, stream>>>(x, xb, nx);
  transpose_w_kernel<<<dim3(N3E / 32, EE / 32), dim3(32, 8), 0, stream>>>(w, wt);
  gemm_qkv_kernel<<<dim3(N3E / 32, BB * SS / 64), 32, 0, stream>>>(xb, wt, qkv);
  rope_kernel<<<(BB * SS * HH * 64 + 255) / 256, 256, 0, stream>>>(qkv);
  vtr_kernel<<<dim3(SS / 32, DD / 32, BB * HH), dim3(32, 8), 0, stream>>>(qkv, vt);
  attn_kernel<<<dim3(SS / 16, BB * HH), 32, 0, stream>>>(qkv, vt, out);
}
